// FCTreeNet_4810363373031
// MI455X (gfx1250) — matrix operand order repaired and checked by reading, not compiled, not run on hardware
//
#include <hip/hip_runtime.h>
#include <hip/hip_bf16.h>

typedef __attribute__((ext_vector_type(16))) __bf16 v16bf;
typedef __attribute__((ext_vector_type(8)))  __bf16 v8bf;
typedef __attribute__((ext_vector_type(8)))  float  v8f;

#define IN_DIM  300
#define IN_PAD  320
#define IMG_DIM 256
#define CAT     556
#define NB      4096
#define NP      16
#define NROWS   (NB * NP)    // 65536
#define XROWS   (NB * 6)     // 24576
#define ASTR    264          // LDS activation row stride (bf16), padded vs 256
#define WSTR    264          // LDS weight row stride (bf16)

// K-group permutation: within each 32-wide K group, storage order is
// [0..7, 16..23, 8..15, 24..31] (swap middle chunks). Involution.
// A-fragment for lane L then reads ONE contiguous 16-element run at
// pos = k0 + (L>=16)*16.
__device__ __forceinline__ int permk(int k) {
    int j = k & 31;
    int ch = j >> 3;                       // chunk 0..3
    int nch = ((ch & 1) << 1) | (ch >> 1); // 0,2,1,3
    return (k & ~31) | (nch << 3) | (j & 7);
}

__device__ __forceinline__ v16bf load_frag(const __bf16* p) {
    v8bf lo = *(const v8bf*)(p);
    v8bf hi = *(const v8bf*)(p + 8);
    return __builtin_shufflevector(lo, hi, 0, 1, 2, 3, 4, 5, 6, 7,
                                   8, 9, 10, 11, 12, 13, 14, 15);
}

union Pack8 { v8bf v; __bf16 b[8]; };
union Pack4 { uint2 u; __bf16 b[4]; };

// ---------------------------------------------------------------------------
// conversion kernels (cold, pure bandwidth): zero-pad + bf16 + K-permute
// ---------------------------------------------------------------------------

// inp (24576 x 300 f32) -> inpb (24576 x 320 bf16, K-permuted, zero-padded)
__global__ __launch_bounds__(256) void convert_inp(const float* __restrict__ inp,
                                                   __bf16* __restrict__ inpb) {
    int idx = blockIdx.x * 256 + threadIdx.x;   // 983040 chunks of 8
    int row = idx / 40;
    int p0 = (idx % 40) * 8;
    int j0 = permk(p0);
    const float* src = inp + (size_t)row * IN_DIM;
    Pack8 pk;
#pragma unroll
    for (int e = 0; e < 8; ++e) {
        int j = j0 + e;
        pk.b[e] = (j < IN_DIM) ? (__bf16)src[j] : (__bf16)0.f;
    }
    *(v8bf*)(inpb + (size_t)row * IN_PAD + p0) = pk.v;
}

// W_fc (300x300 f32) -> Wfcb (320x320 bf16, rows=out col, zero-padded, K-permuted)
__global__ __launch_bounds__(256) void convert_wfc(const float* __restrict__ Wfc,
                                                   __bf16* __restrict__ Wfcb) {
    int idx = blockIdx.x * 256 + threadIdx.x;   // 12800 chunks of 8
    int n = idx / 40;
    int k0 = (idx % 40) * 8;
    const float* src = Wfc + (size_t)(n < IN_DIM ? n : 0) * IN_DIM;
    Pack8 pk;
#pragma unroll
    for (int e = 0; e < 8; ++e) {
        int k = permk(k0) + e;
        pk.b[e] = (n < IN_DIM && k < IN_DIM) ? (__bf16)src[k] : (__bf16)0.f;
    }
    *(v8bf*)(Wfcb + (size_t)n * IN_PAD + k0) = pk.v;
}

// word halves (cols 0..299) of 4 weights -> Wwordb[4][256][320] bf16 zero-padded
__global__ __launch_bounds__(256) void convert_wword(const float* __restrict__ Wl,
                                                     const float* __restrict__ Wm,
                                                     const float* __restrict__ Wg,
                                                     const float* __restrict__ Wr,
                                                     __bf16* __restrict__ out) {
    int idx = blockIdx.x * 256 + threadIdx.x;   // 40960 chunks of 8
    int w = idx / (256 * 40);
    int rem = idx % (256 * 40);
    int o = rem / 40;
    int k0 = (rem % 40) * 8;
    const float* W = (w == 0) ? Wl : (w == 1) ? Wm : (w == 2) ? Wg : Wr;
    const float* src = W + (size_t)o * CAT;
    Pack8 pk;
#pragma unroll
    for (int e = 0; e < 8; ++e) {
        int k = permk(k0) + e;
        pk.b[e] = (k < IN_DIM) ? (__bf16)src[k] : (__bf16)0.f;
    }
    *(v8bf*)(out + ((size_t)w * IMG_DIM + o) * IN_PAD + k0) = pk.v;
}

// feature halves (cols 300..555) of 4 weights -> Wfeatb[4][256][256] bf16
__global__ __launch_bounds__(256) void convert_wfeat(const float* __restrict__ Wl,
                                                     const float* __restrict__ Wm,
                                                     const float* __restrict__ Wg,
                                                     const float* __restrict__ Wr,
                                                     __bf16* __restrict__ out) {
    int idx = blockIdx.x * 256 + threadIdx.x;   // 32768 chunks of 8
    int w = idx / (256 * 32);
    int rem = idx % (256 * 32);
    int o = rem / 32;
    int k0 = (rem % 32) * 8;
    const float* W = (w == 0) ? Wl : (w == 1) ? Wm : (w == 2) ? Wg : Wr;
    const float* src = W + (size_t)o * CAT + 300 + permk(k0);
    Pack8 pk;
#pragma unroll
    for (int e = 0; e < 8; ++e) pk.b[e] = (__bf16)src[e];
    *(v8bf*)(out + ((size_t)w * IMG_DIM + o) * IMG_DIM + k0) = pk.v;
}

// ---------------------------------------------------------------------------
// k1: x = inp @ W_fc.T + b_fc -> xbf (24576 x 320 bf16, K-permuted)
// one wave per 64 rows x 16 cols: load phase (4 A frags live) then WMMA phase
// ---------------------------------------------------------------------------
__global__ __launch_bounds__(32) void fc_gemm(const __bf16* __restrict__ inpb,
                                              const __bf16* __restrict__ Wfcb,
                                              const float* __restrict__ bfc,
                                              __bf16* __restrict__ xout) {
    int job = blockIdx.x;            // 384 row-groups x 20 col-tiles
    int ct = job % 20;
    int rg = job / 20;
    int lane = threadIdx.x;
    int halfsel = lane >> 4;
    int lcol = lane & 15;
    int n = ct * 16 + lcol;

    float bias = (n < IN_DIM) ? bfc[n] : 0.f;
    v8f acc[4];
#pragma unroll
    for (int t = 0; t < 4; ++t)
#pragma unroll
        for (int r = 0; r < 8; ++r) acc[t][r] = bias;

    if (ct < 19) {
        const __bf16* wr = Wfcb + (size_t)n * IN_PAD + halfsel * 16;
        const __bf16* ar = inpb + (size_t)(rg * 64 + lcol) * IN_PAD + halfsel * 16;
#pragma unroll
        for (int ks = 0; ks < 10; ++ks) {
            v16bf bf = load_frag(wr + ks * 32);
            v16bf afr[4];
#pragma unroll
            for (int t = 0; t < 4; ++t)
                afr[t] = load_frag(ar + (size_t)t * 16 * IN_PAD + ks * 32);
#pragma unroll
            for (int t = 0; t < 4; ++t)
                acc[t] = __builtin_amdgcn_wmma_f32_16x16x32_bf16(false, afr[t], false, bf, (short)0, acc[t], false, false);
        }
    }
    int pn = permk(n);
#pragma unroll
    for (int t = 0; t < 4; ++t)
#pragma unroll
        for (int r = 0; r < 8; ++r) {
            int row = rg * 64 + t * 16 + r + halfsel * 8;
            xout[(size_t)row * IN_PAD + pn] = (__bf16)acc[t][r];
        }
}

// ---------------------------------------------------------------------------
// k2: word terms wt[kn][b][n] = x[b,kn,:] @ Wword_kn.T + b_kn[n]
// node->weight idx: 0:root(3) 1:merge(2) 2:middle(1) 3:leaf(0) 4:middle(1) 5:leaf(0)
// ---------------------------------------------------------------------------
__global__ __launch_bounds__(32) void word_gemm(const __bf16* __restrict__ xbf,
                                                const __bf16* __restrict__ Wwordb,
                                                const float* __restrict__ bl,
                                                const float* __restrict__ bm,
                                                const float* __restrict__ bg_,
                                                const float* __restrict__ br,
                                                float* __restrict__ wt) {
    int job = blockIdx.x;            // kn(6) x bg(64) x ct(16) = 6144
    int ct = job & 15;
    int bg = (job >> 4) & 63;
    int kn = job >> 10;

    int widx = (kn == 0) ? 3 : (kn == 1) ? 2 : (kn == 2 || kn == 4) ? 1 : 0;
    const float* bias = (widx == 0) ? bl : (widx == 1) ? bm : (widx == 2) ? bg_ : br;

    int lane = threadIdx.x;
    int halfsel = lane >> 4;
    int lcol = lane & 15;
    int n = ct * 16 + lcol;

    float bb = bias[n];
    v8f acc[4];
#pragma unroll
    for (int t = 0; t < 4; ++t)
#pragma unroll
        for (int r = 0; r < 8; ++r) acc[t][r] = bb;

    const __bf16* wr = Wwordb + ((size_t)widx * IMG_DIM + n) * IN_PAD + halfsel * 16;
    const __bf16* ar = xbf + ((size_t)(bg * 64 + lcol) * 6 + kn) * IN_PAD + halfsel * 16;
#pragma unroll
    for (int ks = 0; ks < 10; ++ks) {
        v16bf bf = load_frag(wr + ks * 32);
        v16bf afr[4];
#pragma unroll
        for (int t = 0; t < 4; ++t)
            afr[t] = load_frag(ar + (size_t)t * 16 * 6 * IN_PAD + ks * 32);
#pragma unroll
        for (int t = 0; t < 4; ++t)
            acc[t] = __builtin_amdgcn_wmma_f32_16x16x32_bf16(false, afr[t], false, bf, (short)0, acc[t], false, false);
    }
#pragma unroll
    for (int t = 0; t < 4; ++t)
#pragma unroll
        for (int r = 0; r < 8; ++r) {
            int b_ = bg * 64 + t * 16 + r + halfsel * 8;
            wt[((size_t)kn * NB + b_) * IMG_DIM + n] = acc[t][r];
        }
}

// ---------------------------------------------------------------------------
// k3: fused tree. Block = 256 threads (8 waves), 64 rows (= 4 batches).
// Activations in LDS (K-permuted); per-stage weight staged in LDS.
// Each wave: 2 col-tile groups x 4 row-tiles; hoisted B frag, load/WMMA phases.
// ---------------------------------------------------------------------------
__global__ __launch_bounds__(256) void tree_kernel(const float* __restrict__ img,
                                                   const float* __restrict__ indicator,
                                                   const __bf16* __restrict__ Wbf,
                                                   const float* __restrict__ wt,
                                                   float* __restrict__ out) {
    extern __shared__ char smem[];
    __bf16* bufI = (__bf16*)smem;           // 64 x ASTR
    __bf16* bufL = bufI + 64 * ASTR;        // 64 x ASTR
    __bf16* bufR = bufL + 64 * ASTR;        // 64 x ASTR
    __bf16* wlds = bufR + 64 * ASTR;        // 256 x WSTR

    const int tid = threadIdx.x;
    const int wave = tid >> 5;
    const int lane = tid & 31;
    const int halfsel = lane >> 4;
    const int lcol = lane & 15;
    const int row0 = blockIdx.x * 64;
    const int bt0 = row0 >> 4;

    // ---- image tile: 64 x 256 f32 -> bf16 LDS (K-permuted positions) ----
#pragma unroll
    for (int it = 0; it < 16; ++it) {
        int idx = tid + it * 256;           // 4096 float4 chunks
        int r = idx >> 6;
        int c4 = (idx & 63) * 4;
        float4 v = *(const float4*)(img + (size_t)(row0 + r) * IMG_DIM + c4);
        Pack4 st;
        st.b[0] = (__bf16)v.x; st.b[1] = (__bf16)v.y;
        st.b[2] = (__bf16)v.z; st.b[3] = (__bf16)v.w;
        *(uint2*)(bufI + r * ASTR + permk(c4)) = st.u;
    }

    auto stage_w = [&](int widx) {
        const __bf16* src = Wbf + (size_t)widx * IMG_DIM * IMG_DIM;
#pragma unroll
        for (int it = 0; it < 32; ++it) {
            int idx = tid + it * 256;       // 8192 chunks of 8 bf16
            int o = idx >> 5;
            int k = (idx & 31) * 8;
            *(v8bf*)(wlds + o * WSTR + k) = *(const v8bf*)(src + (size_t)o * IMG_DIM + k);
        }
    };

    // dstb==nullptr -> final stage, write f32 to global out
    auto gemm_stage = [&](const __bf16* srcb, __bf16* dstb, int node, bool accum, bool useInd) {
        for (int ctj = 0; ctj < 2; ++ctj) {
            int ct = wave + ctj * 8;
            int n = ct * 16 + lcol;
            v8f acc[4];
#pragma unroll
            for (int t = 0; t < 4; ++t) {
                float cv = wt[((size_t)node * NB + bt0 + t) * IMG_DIM + n];  // word+bias
#pragma unroll
                for (int r = 0; r < 8; ++r) acc[t][r] = cv;
            }
            const __bf16* wr = wlds + n * WSTR + halfsel * 16;
            const __bf16* ar = srcb + lcol * ASTR + halfsel * 16;
#pragma unroll
            for (int ks = 0; ks < 8; ++ks) {
                v16bf bf = load_frag(wr + ks * 32);
                v16bf afr[4];
#pragma unroll
                for (int t = 0; t < 4; ++t)
                    afr[t] = load_frag(ar + t * 16 * ASTR + ks * 32);
#pragma unroll
                for (int t = 0; t < 4; ++t)
                    acc[t] = __builtin_amdgcn_wmma_f32_16x16x32_bf16(false, afr[t], false, bf, (short)0, acc[t], false, false);
            }
            if (dstb) {
                int pn = permk(n);
#pragma unroll
                for (int t = 0; t < 4; ++t) {
                    float indv = useInd ? indicator[bt0 + t] : 1.f;
                    __bf16* dp0 = dstb + (t * 16 + halfsel * 8) * ASTR + pn;
#pragma unroll
                    for (int r = 0; r < 8; ++r) {
                        float v = acc[t][r];
                        v = v > 0.f ? v : 0.f;
                        v *= indv;
                        __bf16* dp = dp0 + r * ASTR;
                        if (accum) v += (float)(*dp);
                        *dp = (__bf16)v;
                    }
                }
            } else {
#pragma unroll
                for (int t = 0; t < 4; ++t) {
                    float* op0 = out + (size_t)(row0 + t * 16 + halfsel * 8) * IMG_DIM + n;
#pragma unroll
                    for (int r = 0; r < 8; ++r) {
                        float v = acc[t][r];
                        op0[(size_t)r * IMG_DIM] = v > 0.f ? v : 0.f;
                    }
                }
            }
        }
    };

    stage_w(0);                                 // leaf weight
    __syncthreads();
    gemm_stage(bufI, bufL, 3, false, false);    // leaf-left
    gemm_stage(bufI, bufR, 5, false, false);    // leaf-right
    __syncthreads();
    stage_w(1);                                 // middle weight
    __syncthreads();
    gemm_stage(bufL, bufI, 2, false, false);    // mL -> bufI
    __syncthreads();
    gemm_stage(bufR, bufI, 4, true, true);      // bufI += mR * ind
    __syncthreads();
    stage_w(2);                                 // merge weight
    __syncthreads();
    gemm_stage(bufI, bufL, 1, false, false);    // merge -> bufL
    __syncthreads();
    stage_w(3);                                 // root weight
    __syncthreads();
    gemm_stage(bufL, nullptr, 0, false, false); // root -> global out
}

// ---------------------------------------------------------------------------
extern "C" void kernel_launch(void* const* d_in, const int* in_sizes, int n_in,
                              void* d_out, int out_size, void* d_ws, size_t ws_size,
                              hipStream_t stream) {
    (void)in_sizes; (void)n_in; (void)out_size; (void)ws_size;
    const float* img = (const float*)d_in[0];
    const float* inp = (const float*)d_in[1];
    const float* ind = (const float*)d_in[2];
    const float* Wfc = (const float*)d_in[3];
    const float* bfc = (const float*)d_in[4];
    const float* Wl  = (const float*)d_in[5];
    const float* bl  = (const float*)d_in[6];
    const float* Wm  = (const float*)d_in[7];
    const float* bm  = (const float*)d_in[8];
    const float* Wg  = (const float*)d_in[9];
    const float* bg  = (const float*)d_in[10];
    const float* Wr  = (const float*)d_in[11];
    const float* br  = (const float*)d_in[12];

    char* ws = (char*)d_ws;
    __bf16* Wfeatb = (__bf16*)ws;                       // 524288 B
    __bf16* Wfcb   = (__bf16*)(ws + 524288);            // 204800 B
    __bf16* Wwordb = (__bf16*)(ws + 729088);            // 655360 B
    __bf16* xbf    = (__bf16*)(ws + 1384448);           // 15728640 B
    // overlay: inpb used only before word_gemm; wt reuses the same region
    __bf16* inpb   = (__bf16*)(ws + 17113088);          // 15728640 B
    float*  wt     = (float*)(ws + 17113088);           // 25165824 B (overlays inpb)

    convert_inp  <<<3840, 256, 0, stream>>>(inp, inpb);
    convert_wfc  <<<50,   256, 0, stream>>>(Wfc, Wfcb);
    convert_wword<<<160,  256, 0, stream>>>(Wl, Wm, Wg, Wr, Wwordb);
    convert_wfeat<<<128,  256, 0, stream>>>(Wl, Wm, Wg, Wr, Wfeatb);

    fc_gemm<<<384 * 20, 32, 0, stream>>>(inpb, Wfcb, bfc, xbf);
    word_gemm<<<6 * 64 * 16, 32, 0, stream>>>(xbf, Wwordb, bl, bm, bg, br, wt);

    size_t lds_bytes = (size_t)(3 * 64 * ASTR + 256 * WSTR) * sizeof(__bf16);  // 236544 B
    tree_kernel<<<NROWS / 64, 256, lds_bytes, stream>>>(img, ind, Wfeatb, wt, (float*)d_out);
}
